// TrajectoryRasterizer_22265110462774
// MI455X (gfx1250) — matrix operand order repaired and checked by reading, not compiled, not run on hardware
//
#include <hip/hip_runtime.h>
#include <hip/hip_bf16.h>

// Problem dims (from reference): traj (4,16,32,2), lat (181,), lon (360,)
#define BDIM     4
#define TDIM     16
#define NBT      (BDIM * TDIM)   // 64 images
#define NPTS     32              // Gaussians per image (WMMA K dim!)
#define H        181
#define W        360
#define HP       192             // H padded to 12 tiles of 16
#define WP       368             // W padded to 23 tiles of 16
#define TILES_I  12
#define TILES_J  23
#define NTILES   (TILES_I * TILES_J)   // 276 tiles per image
#define EDGE_ROWS (H - (TILES_I - 1) * 16)   // 5 valid rows in bottom tiles
#define LSTRIDE  40              // LDS row stride in halves (80B: bank-conflict-free, 16B aligned)

typedef __attribute__((ext_vector_type(16))) _Float16 v16h;
typedef __attribute__((ext_vector_type(8)))  _Float16 v8h;
typedef __attribute__((ext_vector_type(8)))  float    v8f;

// ---------------------------------------------------------------------------
// Kernel 1: one workgroup per (b,t) image.
//   Phase 1: build separable Gaussian tables A[i][o], B[j][o] in LDS (f16).
//   Phase 2: heat = A(181x32) x B^T(32x360) via v_wmma_f32_16x16x32_f16,
//            one WMMA per 16x16 output tile (K=32 exactly).
//   Tile coords are forced uniform via readfirstlane so interior tiles get a
//   pure scalar branch and 8 unconditional stores; each edge tile pays a
//   single exec-mask region.
// ---------------------------------------------------------------------------
__global__ __launch_bounds__(256) void traj_heat_wmma(
    const float* __restrict__ traj,   // (b,t,o,2)
    const float* __restrict__ lat,    // (181,)
    const float* __restrict__ lon,    // (360,)
    float* __restrict__ out,          // (b,t,181,360) unnormalized
    unsigned int* __restrict__ gmax)  // (64,) per-image max as uint bits
{
    __shared__ _Float16 Ah[HP * LSTRIDE];   // A[i][o] : exp(-dlat^2/2s^2)
    __shared__ _Float16 Bh[WP * LSTRIDE];   // B[j][o] : exp(-dlon^2/2s^2)
    __shared__ float plat[NPTS], plon[NPTS];

    const int bt  = blockIdx.x;
    const int tid = threadIdx.x;
    const float coef = -0.5f / (2.5f * 2.5f + 1e-8f);

    if (tid < NPTS) {
        plat[tid] = traj[(bt * NPTS + tid) * 2 + 0];
        plon[tid] = traj[(bt * NPTS + tid) * 2 + 1];
    }
    __syncthreads();

    // Phase 1: fill exp tables (padded rows/cols set to 0 -> contribute nothing)
    for (int idx = tid; idx < HP * NPTS; idx += 256) {
        int i = idx >> 5, o = idx & 31;
        float v = 0.0f;
        if (i < H) { float d = lat[i] - plat[o]; v = __expf(coef * d * d); }
        Ah[i * LSTRIDE + o] = (_Float16)v;
    }
    for (int idx = tid; idx < WP * NPTS; idx += 256) {
        int j = idx >> 5, o = idx & 31;
        float v = 0.0f;
        if (j < W) { float d = lon[j] - plon[o]; v = __expf(coef * d * d); }
        Bh[j * LSTRIDE + o] = (_Float16)v;
    }
    __syncthreads();

    const int  lane   = tid & 31;
    const int  lane15 = lane & 15;
    const bool hihalf = lane >= 16;
    // Wave index as a compiler-provable scalar.
    const int  wave   = __builtin_amdgcn_readfirstlane(tid >> 5);
    float lmax = 0.0f;

    // Phase 2: scalar tile walk; EXEC is all-ones at every WMMA.
    int ti = 0, tj = wave;                       // wave < 8 < TILES_J
    for (int tile = wave; tile < NTILES; tile += 8) {
        // A fragment (16x32 f16, ISA 7.12.2): row M = lane15.
        //   lanes 0-15 : halves[0..7]=K0..7,  halves[8..15]=K16..23
        //   lanes 16-31: halves[0..7]=K8..15, halves[8..15]=K24..31
        const _Float16* arow = &Ah[(ti * 16 + lane15) * LSTRIDE];
        const int ak = hihalf ? 8 : 0;
        v8h alo = *(const v8h*)(arow + ak);        // ds_load_b128
        v8h ahi = *(const v8h*)(arow + ak + 16);   // ds_load_b128

        // B fragment (32x16 f16): column N = lane15.
        //   lanes 0-15 : halves[0..7]=K0..7,  halves[8..15]=K16..23
        //   lanes 16-31: halves[0..7]=K8..15, halves[8..15]=K24..31
        const _Float16* bcol = &Bh[(tj * 16 + lane15) * LSTRIDE];
        const int bk = hihalf ? 8 : 0;
        v8h blo = *(const v8h*)(bcol + bk);
        v8h bhi = *(const v8h*)(bcol + bk + 16);

        v16h a, b;
#pragma unroll
        for (int e = 0; e < 8; ++e) {
            a[e] = alo[e]; a[e + 8] = ahi[e];
            b[e] = blo[e]; b[e + 8] = bhi[e];
        }

        v8f c = {};
        // D = A x B + 0 : one matrix-core op per 16x16 output tile (K=32)
        c = __builtin_amdgcn_wmma_f32_16x16x32_f16(
                /*neg_a=*/false, a, /*neg_b=*/false, b,
                /*c_mod=*/(short)0, c, /*reuse_a=*/false, /*reuse_b=*/false);

#pragma unroll
        for (int r = 0; r < 8; ++r)
            lmax = fmaxf(lmax, c[r]);   // padded entries are exactly 0 -> harmless

        // C/D layout: VGPR r -> M = mb + r, N = lane15.
        const int gj = tj * 16 + lane15;                 // column (vector)
        const int mb = ti * 16 + (hihalf ? 8 : 0);       // first row of half
        float* __restrict__ orow = out + ((size_t)(bt * H + mb)) * W + gj;

        if (ti < TILES_I - 1) {
            if (tj < TILES_J - 1) {
                // Interior tile: scalar branch, 8 unconditional stores.
#pragma unroll
                for (int r = 0; r < 8; ++r)
                    orow[r * W] = c[r];
            } else {
                // Right edge: single per-lane column guard around all 8 stores.
                if (gj < W) {
#pragma unroll
                    for (int r = 0; r < 8; ++r)
                        orow[r * W] = c[r];
                }
            }
        } else {
            // Bottom tiles: valid rows 0..EDGE_ROWS-1 live only in lo-half
            // lanes -> one exec region, EDGE_ROWS stores.
            if (!hihalf && gj < W) {
#pragma unroll
                for (int r = 0; r < EDGE_ROWS; ++r)
                    orow[r * W] = c[r];
            }
        }

        // Advance tile coords without division (scalar).
        tj += 8;
        if (tj >= TILES_J) { tj -= TILES_J; ++ti; }
    }

    // All heat values are > 0, so uint compare == float compare.
    atomicMax(&gmax[bt], __float_as_uint(lmax));
}

// ---------------------------------------------------------------------------
// Kernel 2: normalize each image by its max (+eps). Pure bandwidth pass.
// ---------------------------------------------------------------------------
__global__ __launch_bounds__(256) void traj_heat_norm(
    float* __restrict__ out, const unsigned int* __restrict__ gmax, int total)
{
    int idx = blockIdx.x * 256 + threadIdx.x;
    if (idx >= total) return;
    int bt = idx / (H * W);
    float m = __uint_as_float(gmax[bt]);
    out[idx] = out[idx] / (m + 1e-8f);
}

extern "C" void kernel_launch(void* const* d_in, const int* in_sizes, int n_in,
                              void* d_out, int out_size, void* d_ws, size_t ws_size,
                              hipStream_t stream) {
    const float* traj = (const float*)d_in[0];   // 4*16*32*2 f32
    const float* lat  = (const float*)d_in[1];   // 181 f32
    const float* lon  = (const float*)d_in[2];   // 360 f32
    float* out        = (float*)d_out;           // 4*16*181*360 f32
    unsigned int* gmx = (unsigned int*)d_ws;     // 64 uints

    // Zero the per-image max slots (capture-safe stream memset).
    hipMemsetAsync(d_ws, 0, NBT * sizeof(unsigned int), stream);

    traj_heat_wmma<<<NBT, 256, 0, stream>>>(traj, lat, lon, out, gmx);

    const int total = NBT * H * W;
    traj_heat_norm<<<(total + 255) / 256, 256, 0, stream>>>(out, gmx, total);
}
